// SKAttention_3607772529147
// MI455X (gfx1250) — matrix operand order repaired and checked by reading, not compiled, not run on hardware
//
#include <hip/hip_runtime.h>
#include <hip/hip_bf16.h>

typedef __attribute__((ext_vector_type(16))) _Float16 v16h;
typedef __attribute__((ext_vector_type(8)))  float    v8f;

#define BS   128
#define CH   512
#define DD   32
#define NP   16   // patches

// ---------------------------------------------------------------------------
// Kernel 1: S[p][b][c] = mean_{4x4}( (x+x1) patch p of image (b,c) )
// thread t -> (b, rowgroup rg, c); reads 4 rows x 16 cols (float4), makes
// 4 patch sums (p = 4*rg + j). Writes coalesced over c.
// ---------------------------------------------------------------------------
__global__ void sk_reduce_kernel(const float* __restrict__ x,
                                 const float* __restrict__ x1,
                                 float* __restrict__ S) {
    int t  = blockIdx.x * blockDim.x + threadIdx.x;   // BS*4*CH threads
    int c  = t % CH;
    int rg = (t / CH) & 3;
    int b  = t / (CH * 4);

    const float4* px  = (const float4*)(x  + ((size_t)(b * CH + c)) * 256 + rg * 64);
    const float4* px1 = (const float4*)(x1 + ((size_t)(b * CH + c)) * 256 + rg * 64);

    float sums[4] = {0.f, 0.f, 0.f, 0.f};
    #pragma unroll
    for (int row = 0; row < 4; ++row) {
        #pragma unroll
        for (int j = 0; j < 4; ++j) {
            float4 a = px[row * 4 + j];
            float4 d = px1[row * 4 + j];
            sums[j] += (a.x + d.x) + (a.y + d.y) + (a.z + d.z) + (a.w + d.w);
        }
    }
    #pragma unroll
    for (int j = 0; j < 4; ++j) {
        // S layout: [p][b][c], p = 4*rg + j
        S[(size_t)(4 * rg + j) * (BS * CH) + (size_t)b * CH + c] = sums[j] * 0.0625f;
    }
}

// ---------------------------------------------------------------------------
// Kernel 2 (WMMA): gate computation.
//   Z[m, d]      = S[m, :] . fc_w[d, :]  + fc_b[d]        (M=2048, K=512, N=32)
//   l_k[m, c]    = Z[m, :] . fcs_w[k, c, :] + fcs_b[k, c] (M=2048, K=32, N=512)
//   attn0[m, c]  = 1 / (1 + exp(l1 - l0))                 (softmax over k=2)
// One wave per 16-row M tile (p constant per tile since 128 % 16 == 0).
// v_wmma_f32_16x16x32_f16, f32 accumulate.
// Fragment layouts (A/B k order is this code's contiguous choice, not the ISA table order):
//   A(16x32 f16):  lane m = lane&15, hi = lane>>4; half e -> K = 16*hi + e
//   B(32x16 f16):  lane n = lane&15;               half e -> K = 16*hi + e
//   C/D(16x16 f32): vgpr r -> M = r + 8*hi, N = lane&15
// ---------------------------------------------------------------------------
__global__ void sk_attn_kernel(const float* __restrict__ S,
                               const float* __restrict__ fc_w,
                               const float* __restrict__ fc_b,
                               const float* __restrict__ fcs_w,
                               const float* __restrict__ fcs_b,
                               float* __restrict__ attn0) {
    __shared__ float zbuf[8][16][DD];   // per-wave Z tile re-pack buffer (16 KB)

    const int wave = threadIdx.x >> 5;
    const int lane = threadIdx.x & 31;
    const int tile = blockIdx.x * 8 + wave;   // 0..127
    const int m0   = tile * 16;
    const int mrow = lane & 15;
    const int hi   = lane >> 4;
    const int n16  = lane & 15;

    // ---- GEMM 1: Z = S * fc_w^T  (two 16-wide N tiles -> d = 0..31) ----
    v8f accZ0 = {};
    v8f accZ1 = {};
    for (int kb = 0; kb < 16; ++kb) {
        const int kbase = kb * 32;
        v16h a;
        #pragma unroll
        for (int e = 0; e < 16; ++e) {
            int koff = 16 * hi + e;
            a[e] = (_Float16)S[(size_t)(m0 + mrow) * CH + kbase + koff];
        }
        v16h b0, b1;
        #pragma unroll
        for (int e = 0; e < 16; ++e) {
            int k = kbase + 16 * hi + e;
            b0[e] = (_Float16)fc_w[(size_t)n16 * CH + k];          // B(k, n=d)   = fc_w[d, k]
            b1[e] = (_Float16)fc_w[(size_t)(n16 + 16) * CH + k];
        }
        accZ0 = __builtin_amdgcn_wmma_f32_16x16x32_f16(false, a, false, b0,
                                                       (short)0, accZ0, false, false);
        accZ1 = __builtin_amdgcn_wmma_f32_16x16x32_f16(false, a, false, b1,
                                                       (short)0, accZ1, false, false);
    }

    // Bias + re-pack Z (D layout) -> A2 fragment (A layout) via LDS.
    const float bias0 = fc_b[n16];
    const float bias1 = fc_b[n16 + 16];
    #pragma unroll
    for (int r = 0; r < 8; ++r) {
        zbuf[wave][r + 8 * hi][n16]      = accZ0[r] + bias0;
        zbuf[wave][r + 8 * hi][n16 + 16] = accZ1[r] + bias1;
    }
    __syncthreads();

    v16h a2;
    #pragma unroll
    for (int e = 0; e < 16; ++e) {
        int koff = 16 * hi + e;   // covers d = 0..31 exactly
        a2[e] = (_Float16)zbuf[wave][mrow][koff];
    }

    // ---- GEMM 2 + softmax gate: N = 512 channels, K = 32 (single step) ----
    for (int nt = 0; nt < 32; ++nt) {
        const int N0 = nt * 16;
        v16h bA, bB;
        #pragma unroll
        for (int e = 0; e < 16; ++e) {
            int kd = 16 * hi + e;
            bA[e] = (_Float16)fcs_w[((size_t)(0 * CH + N0 + n16)) * DD + kd];
            bB[e] = (_Float16)fcs_w[((size_t)(1 * CH + N0 + n16)) * DD + kd];
        }
        v8f c0 = {};
        v8f c1 = {};
        c0 = __builtin_amdgcn_wmma_f32_16x16x32_f16(false, a2, false, bA,
                                                    (short)0, c0, false, false);
        c1 = __builtin_amdgcn_wmma_f32_16x16x32_f16(false, a2, false, bB,
                                                    (short)0, c1, false, false);
        const int   cc  = N0 + n16;
        const float fb0 = fcs_b[cc];
        const float fb1 = fcs_b[CH + cc];
        #pragma unroll
        for (int r = 0; r < 8; ++r) {
            float l0 = c0[r] + fb0;
            float l1 = c1[r] + fb1;
            float g  = 1.0f / (1.0f + __expf(l1 - l0));   // softmax over 2 branches
            attn0[(size_t)(m0 + r + 8 * hi) * CH + cc] = g;
        }
    }
}

// ---------------------------------------------------------------------------
// Kernel 3: outputs. One thread per output float4 (4 cols = one patch-column j).
//   out1[b,ch,R,C] = attn0[p,b,ch]       * x [b,ch,R+4,C] + x [p,ch,R+4,C]
//   out2[b,ch,R,C] = (1 - attn0[p,b,ch]) * x1[b,ch,R+4,C] + x1[p,ch,R+4,C]
//   p = 4*(R/4 + 1) + C/4   (row-group 0 dropped by _cat; batch index p in extra)
// Fully coalesced loads/stores over t.
// ---------------------------------------------------------------------------
__global__ void sk_out_kernel(const float* __restrict__ x,
                              const float* __restrict__ x1,
                              const float* __restrict__ attn0,
                              float* __restrict__ out) {
    int t = blockIdx.x * blockDim.x + threadIdx.x;    // BS*CH*12*4 threads
    int j   = t & 3;
    int R   = (t >> 2) % 12;
    int chb = t / 48;
    int ch  = chb % CH;
    int b   = chb / CH;

    int p   = 4 * (R / 4 + 1) + j;
    int row = R + 4;

    const float4* x4  = (const float4*)x;
    const float4* x14 = (const float4*)x1;
    size_t imgBase = (size_t)(b * CH + ch) * 64;   // 256 floats / 4
    size_t exBase  = (size_t)(p * CH + ch) * 64;
    size_t off     = (size_t)row * 4 + j;

    float4 xr = x4 [imgBase + off];
    float4 ex = x4 [exBase  + off];
    float4 yr = x14[imgBase + off];
    float4 ey = x14[exBase  + off];

    float a0 = attn0[((size_t)p * BS + b) * CH + ch];
    float a1 = 1.0f - a0;

    float4 o1, o2;
    o1.x = a0 * xr.x + ex.x;  o1.y = a0 * xr.y + ex.y;
    o1.z = a0 * xr.z + ex.z;  o1.w = a0 * xr.w + ex.w;
    o2.x = a1 * yr.x + ey.x;  o2.y = a1 * yr.y + ey.y;
    o2.z = a1 * yr.z + ey.z;  o2.w = a1 * yr.w + ey.w;

    float4* out1 = (float4*)out;
    float4* out2 = (float4*)(out + (size_t)BS * CH * 12 * 16);
    size_t oidx = ((size_t)(b * CH + ch) * 12 + R) * 4 + j;
    out1[oidx] = o1;
    out2[oidx] = o2;
}

// ---------------------------------------------------------------------------
extern "C" void kernel_launch(void* const* d_in, const int* in_sizes, int n_in,
                              void* d_out, int out_size, void* d_ws, size_t ws_size,
                              hipStream_t stream) {
    const float* x     = (const float*)d_in[0];
    const float* x1    = (const float*)d_in[1];
    const float* fc_w  = (const float*)d_in[2];
    const float* fc_b  = (const float*)d_in[3];
    const float* fcs_w = (const float*)d_in[4];
    const float* fcs_b = (const float*)d_in[5];
    float* out = (float*)d_out;

    // scratch: S (16*128*512 f32 = 4 MiB) then attn0 (4 MiB)
    float* S     = (float*)d_ws;
    float* attn0 = S + (size_t)NP * BS * CH;

    sk_reduce_kernel<<<(BS * CH * 4) / 256, 256, 0, stream>>>(x, x1, S);
    sk_attn_kernel<<<16, 256, 0, stream>>>(S, fc_w, fc_b, fcs_w, fcs_b, attn0);
    sk_out_kernel<<<(BS * CH * 12 * 4) / 256, 256, 0, stream>>>(x, x1, attn0, out);
}
